// FlashDeepseekAttention_2585570312829
// MI455X (gfx1250) — matrix operand order repaired and checked by reading, not compiled, not run on hardware
//
#include <hip/hip_runtime.h>
#include <hip/hip_bf16.h>
#include <cstdint>

// ---- problem constants (fixed by the reference) ----
#define H_    32
#define HKV_  32
#define D_    128
#define HID_  4096
#define B_    4
#define S_    1024
#define T_    (B_ * S_)
#define NUM_BLOCKS_ 64
#define BLOCK_SIZE_ 128
#define QKV_N ((H_ + 2 * HKV_) * D_)   // 12288
#define SCALE_ 0.08838834764831845f    // 1/sqrt(128)

typedef __attribute__((ext_vector_type(16))) __bf16 v16bf;
typedef __attribute__((ext_vector_type(2)))  __bf16 v2bf;
typedef __attribute__((ext_vector_type(8)))  float  v8f;
typedef __attribute__((ext_vector_type(4)))  float  f4;
typedef __attribute__((ext_vector_type(4)))  int    v4i;

union Frag16 { v16bf v; unsigned u[8]; };

// ---- float -> bf16 via compiler fptrunc (backend selects native cvt) ----
static __device__ __forceinline__ unsigned pack2(float a, float b) {
    union { v2bf v; unsigned u; } cv;
    cv.v[0] = (__bf16)a;
    cv.v[1] = (__bf16)b;
    return cv.u;
}
static __device__ __forceinline__ unsigned short f2bf(float f) {
    union { __bf16 h; unsigned short u; } cv;
    cv.h = (__bf16)f;
    return cv.u;
}

static __device__ __forceinline__ v8f vzero() {
    v8f z;
#pragma unroll
    for (int i = 0; i < 8; ++i) z[i] = 0.f;
    return z;
}
// A-fragment (16x32, 16-bit): lane m=l&15; VGPR v holds K pair starting at:
static __device__ __forceinline__ int akmap(int v, int kh) {
    return kh * 16 + 2 * v;
}
static __device__ __forceinline__ v8f wmma_bf16(const Frag16& a, const Frag16& b, v8f c) {
    return __builtin_amdgcn_wmma_f32_16x16x32_bf16(false, a.v, false, b.v, (short)0, c, false, false);
}

// ---- CDNA5 async global->LDS copy (16B per lane) + counters ----
typedef __attribute__((address_space(1))) v4i gas_v4i;   // global
typedef __attribute__((address_space(3))) v4i las_v4i;   // LDS

static __device__ __forceinline__ void async_copy16(const float* g, float* l) {
#if __has_builtin(__builtin_amdgcn_global_load_async_to_lds_b128)
    __builtin_amdgcn_global_load_async_to_lds_b128(
        (gas_v4i*)(uintptr_t)g, (las_v4i*)(uintptr_t)l, 0, 0);
#else
    asm volatile("global_load_async_to_lds_b128 %0, %1, off"
                 :: "v"((unsigned)(uintptr_t)l), "v"(g) : "memory");
#endif
}
static __device__ __forceinline__ void wait_async0() {
#if __has_builtin(__builtin_amdgcn_s_wait_asynccnt)
    __builtin_amdgcn_s_wait_asynccnt(0);
#else
    asm volatile("s_wait_asynccnt 0" ::: "memory");
#endif
}
static __device__ __forceinline__ void wait_ds0() {
    asm volatile("s_wait_dscnt 0" ::: "memory");
}

// =====================================================================
// Tiled GEMM: C[M,N] = A[M,K] * B[K,N], fp32 in/out, bf16 WMMA math.
// 128x128 C tile, 256 threads (8 waves), 2x4 16x16 tiles per wave,
// double-buffered LDS (packed-bf16 pairs), 128-bit global loads.
// =====================================================================
__global__ __launch_bounds__(256) void gemm_bf16_wmma(
    const float* __restrict__ A, const float* __restrict__ B, float* __restrict__ C,
    int M, int N, int K, int lda, int ldb, int ldc)
{
    __shared__ unsigned sAu[2][128][16];   // [m][k-pair] packed bf16x2
    __shared__ unsigned sBu[2][128][16];   // [n][k-pair] packed bf16x2 (B transposed)

    const int tid  = threadIdx.x;
    const int lane = tid & 31;
    const int wave = tid >> 5;      // 0..7
    const int wm   = wave >> 1;     // 0..3 -> rows 32*wm
    const int wn   = wave & 1;      // 0..1 -> cols 64*wn
    const int mg   = lane & 15;
    const int kh   = lane >> 4;
    const int bm   = blockIdx.y * 128;
    const int bn   = blockIdx.x * 128;

    v8f acc[2][4];
#pragma unroll
    for (int i = 0; i < 2; ++i)
#pragma unroll
        for (int j = 0; j < 4; ++j) acc[i][j] = vzero();

    f4 ra[4], rb[2][2];

    // ---- prologue: load first K-tile into buffer 0 ----
#pragma unroll
    for (int j = 0; j < 4; ++j) {
        int q = tid + j * 256, m = q >> 3, kq = (q & 7) * 4;
        ra[j] = *(const f4*)&A[(size_t)(bm + m) * lda + kq];
    }
#pragma unroll
    for (int j = 0; j < 2; ++j) {
        int g = tid + j * 256, kp = g & 15, nq = g >> 4;
        rb[j][0] = *(const f4*)&B[(size_t)(2 * kp) * ldb + bn + 4 * nq];
        rb[j][1] = *(const f4*)&B[(size_t)(2 * kp + 1) * ldb + bn + 4 * nq];
    }
#pragma unroll
    for (int j = 0; j < 4; ++j) {
        int q = tid + j * 256, m = q >> 3, kq = (q & 7) * 4;
        sAu[0][m][kq >> 1]       = pack2(ra[j][0], ra[j][1]);
        sAu[0][m][(kq >> 1) + 1] = pack2(ra[j][2], ra[j][3]);
    }
#pragma unroll
    for (int j = 0; j < 2; ++j) {
        int g = tid + j * 256, kp = g & 15, nq = g >> 4;
#pragma unroll
        for (int nn = 0; nn < 4; ++nn)
            sBu[0][4 * nq + nn][kp] = pack2(rb[j][0][nn], rb[j][1][nn]);
    }

    int p = 0;
    for (int k0 = 0; k0 < K; k0 += 32) {
        __syncthreads();
        const bool next = (k0 + 32 < K);
        if (next) {  // prefetch next tile into registers while WMMAs run
            const int kn = k0 + 32;
#pragma unroll
            for (int j = 0; j < 4; ++j) {
                int q = tid + j * 256, m = q >> 3, kq = (q & 7) * 4;
                ra[j] = *(const f4*)&A[(size_t)(bm + m) * lda + kn + kq];
            }
#pragma unroll
            for (int j = 0; j < 2; ++j) {
                int g = tid + j * 256, kp = g & 15, nq = g >> 4;
                rb[j][0] = *(const f4*)&B[(size_t)(kn + 2 * kp) * ldb + bn + 4 * nq];
                rb[j][1] = *(const f4*)&B[(size_t)(kn + 2 * kp + 1) * ldb + bn + 4 * nq];
            }
        }

        Frag16 af[2], bf[4];
#pragma unroll
        for (int mi = 0; mi < 2; ++mi) {
            const unsigned* row = &sAu[p][wm * 32 + mi * 16 + mg][0];
#pragma unroll
            for (int v = 0; v < 8; ++v)
                af[mi].u[v] = row[kh * 8 + v];
        }
#pragma unroll
        for (int ni = 0; ni < 4; ++ni) {
            const unsigned* row = &sBu[p][wn * 64 + ni * 16 + mg][0];
#pragma unroll
            for (int v = 0; v < 8; ++v)
                bf[ni].u[v] = row[kh * 8 + v];
        }
#pragma unroll
        for (int mi = 0; mi < 2; ++mi)
#pragma unroll
            for (int ni = 0; ni < 4; ++ni)
                acc[mi][ni] = wmma_bf16(af[mi], bf[ni], acc[mi][ni]);

        if (next) {  // stage next tile into the other buffer
#pragma unroll
            for (int j = 0; j < 4; ++j) {
                int q = tid + j * 256, m = q >> 3, kq = (q & 7) * 4;
                sAu[p ^ 1][m][kq >> 1]       = pack2(ra[j][0], ra[j][1]);
                sAu[p ^ 1][m][(kq >> 1) + 1] = pack2(ra[j][2], ra[j][3]);
            }
#pragma unroll
            for (int j = 0; j < 2; ++j) {
                int g = tid + j * 256, kp = g & 15, nq = g >> 4;
#pragma unroll
                for (int nn = 0; nn < 4; ++nn)
                    sBu[p ^ 1][4 * nq + nn][kp] = pack2(rb[j][0][nn], rb[j][1][nn]);
            }
        }
        p ^= 1;
    }

    // C layout: VGPR r -> row r + 8*(lane>=16); col = lane&15
    const int rowhi = kh * 8;
#pragma unroll
    for (int mi = 0; mi < 2; ++mi)
#pragma unroll
        for (int ni = 0; ni < 4; ++ni) {
            int gm = bm + wm * 32 + mi * 16 + rowhi;
            int gn = bn + wn * 64 + ni * 16 + mg;
#pragma unroll
            for (int r = 0; r < 8; ++r)
                C[(size_t)(gm + r) * ldc + gn] = acc[mi][ni][r];
        }
}

// =====================================================================
// RoPE (q in-place) + scatter roped k / raw v into the paged caches.
// =====================================================================
__global__ __launch_bounds__(128) void rope_scatter_kernel(
    float* __restrict__ qkv, const float* __restrict__ cosp, const float* __restrict__ sinp,
    const int* __restrict__ slots, float* __restrict__ kc, float* __restrict__ vc)
{
    const int t = blockIdx.x;
    const int d = threadIdx.x;
    const int slot = slots[t];
    float c = 0.f, s = 0.f;
    if (d < 64) { c = cosp[t * 64 + d]; s = sinp[t * 64 + d]; }
    float* row = qkv + (size_t)t * QKV_N;

    if (d < 64) {
#pragma unroll 4
        for (int h = 0; h < H_; ++h) {
            float* qh = row + h * D_;
            float x1 = qh[d], x2 = qh[d + 64];
            qh[d]      = x1 * c - x2 * s;
            qh[d + 64] = x2 * c + x1 * s;
        }
    }
#pragma unroll 4
    for (int h = 0; h < HKV_; ++h) {
        const size_t crow = ((size_t)slot * HKV_ + h) * D_;
        if (d < 64) {
            const float* kh = row + H_ * D_ + h * D_;
            float x1 = kh[d], x2 = kh[d + 64];
            kc[crow + d]      = x1 * c - x2 * s;
            kc[crow + d + 64] = x2 * c + x1 * s;
        }
        vc[crow + d] = row[H_ * D_ + HKV_ * D_ + h * D_ + d];
    }
}

// =====================================================================
// Causal flash attention. Block = 4 waves = 64 query rows of one (b,h).
// K/V chunks (32 keys x 128) stream into LDS via async-to-LDS copies
// shared by all 4 waves; QK^T and PV on WMMA bf16; online softmax.
// =====================================================================
__global__ __launch_bounds__(128) void attn_kernel(
    const float* __restrict__ qkv, const float* __restrict__ kc, const float* __restrict__ vc,
    const int* __restrict__ block_tables, float* __restrict__ attn)
{
    const int qgrp = blockIdx.x, h = blockIdx.y, b = blockIdx.z;
    const int tid = threadIdx.x;
    const int wave = tid >> 5, lane = tid & 31;
    const int qtile = qgrp * 4 + wave;
    const int mg = lane & 15, kh = lane >> 4, rowhi = kh * 8;

    __shared__ float sK[32][128];              // 16 KB: 32 keys x D
    __shared__ float sV[32][128];              // 16 KB
    __shared__ unsigned short sP[4][16][32];   // per-wave P tile (bf16)

    // --- Q fragments: 4x (16x32 bf16) covering D=128 ---
    Frag16 qf[4];
    {
        const int t = b * S_ + qtile * 16 + mg;
        const float* qrow = qkv + (size_t)t * QKV_N + h * D_;
#pragma unroll
        for (int f = 0; f < 4; ++f)
#pragma unroll
            for (int v = 0; v < 8; ++v) {
                int k = f * 32 + akmap(v, kh);
                qf[f].u[v] = pack2(qrow[k], qrow[k + 1]);
            }
    }

    v8f o[8];
#pragma unroll
    for (int i = 0; i < 8; ++i) o[i] = vzero();
    float mrow[8], lrow[8];
#pragma unroll
    for (int r = 0; r < 8; ++r) { mrow[r] = -3.0e38f; lrow[r] = 0.f; }

    const int my_end = qtile * 16 + 16;    // causal limit for this wave
    const int nch = 2 * qgrp + 2;          // chunks for the largest qtile in block
    const int krow = tid >> 2;             // 0..31: key row this thread copies
    const int kseg = (tid & 3) * 32;       // 128B segment within the row

    for (int ch = 0; ch < nch; ++ch) {
        const int base = ch * 32;

        // --- async stage K/V chunk into LDS (all 128 threads cooperate) ---
        {
            const int key = base + krow;   // chunk is 32-aligned -> single page
            const int blk = block_tables[b * (S_ / BLOCK_SIZE_) + (key >> 7)];
            const size_t crow = ((size_t)blk * BLOCK_SIZE_ + (key & 127)) * HKV_ + h;
            const float* gK = kc + crow * D_ + kseg;
            const float* gV = vc + crow * D_ + kseg;
            float* lK = &sK[krow][kseg];
            float* lV = &sV[krow][kseg];
#pragma unroll
            for (int j = 0; j < 8; ++j) async_copy16(gK + j * 4, lK + j * 4);
#pragma unroll
            for (int j = 0; j < 8; ++j) async_copy16(gV + j * 4, lV + j * 4);
        }
        wait_async0();
        __syncthreads();

        if (base < my_end) {   // uniform per wave
            // --- scores: two 16x16 C-frags (keys base..+15, +16..+31) ---
            v8f sc[2];
#pragma unroll
            for (int half = 0; half < 2; ++half) {
                const float* krw = &sK[half * 16 + mg][0];
                v8f a = vzero();
#pragma unroll
                for (int f = 0; f < 4; ++f) {
                    Frag16 bfk;
#pragma unroll
                    for (int v = 0; v < 8; ++v) {
                        int kd = f * 32 + kh * 16 + 2 * v;
                        bfk.u[v] = pack2(krw[kd], krw[kd + 1]);
                    }
                    a = wmma_bf16(qf[f], bfk, a);
                }
                sc[half] = a;
            }

            // --- online softmax (row r+rowhi lives on this 16-lane group) ---
            float p0[8], p1[8], alpha[8];
#pragma unroll
            for (int r = 0; r < 8; ++r) {
                const int qi = qtile * 16 + rowhi + r;
                float v0 = sc[0][r] * SCALE_;
                float v1 = sc[1][r] * SCALE_;
                if (base + mg > qi)      v0 = -3.0e38f;
                if (base + 16 + mg > qi) v1 = -3.0e38f;
                float cm = fmaxf(v0, v1);
#pragma unroll
                for (int i = 1; i < 16; i <<= 1) cm = fmaxf(cm, __shfl_xor(cm, i, 32));
                const float mn = fmaxf(mrow[r], cm);
                const float al = __expf(mrow[r] - mn);
                const float e0 = __expf(v0 - mn), e1 = __expf(v1 - mn);
                float rs = e0 + e1;
#pragma unroll
                for (int i = 1; i < 16; i <<= 1) rs += __shfl_xor(rs, i, 32);
                lrow[r] = lrow[r] * al + rs;
                mrow[r] = mn;
                alpha[r] = al; p0[r] = e0; p1[r] = e1;
            }
#pragma unroll
            for (int f = 0; f < 8; ++f)
#pragma unroll
                for (int r = 0; r < 8; ++r) o[f][r] *= alpha[r];

            // --- relayout P (C-frag) -> A-frag via wave-private LDS ---
#pragma unroll
            for (int r = 0; r < 8; ++r) {
                sP[wave][rowhi + r][mg]      = f2bf(p0[r]);
                sP[wave][rowhi + r][16 + mg] = f2bf(p1[r]);
            }
            wait_ds0();   // wave-local ordering; LDS ops of one wave are in-order
            Frag16 pf;
#pragma unroll
            for (int v = 0; v < 8; ++v)
                pf.u[v] = *(const unsigned*)&sP[wave][mg][akmap(v, kh)];

            // --- PV: 8 WMMAs over the 128 output dims ---
#pragma unroll
            for (int dt = 0; dt < 8; ++dt) {
                Frag16 vf;
#pragma unroll
                for (int v = 0; v < 8; ++v) {
                    const int kk = kh * 16 + 2 * v;
                    const int d = dt * 16 + mg;
                    vf.u[v] = pack2(sV[kk][d], sV[kk + 1][d]);
                }
                o[dt] = wmma_bf16(pf, vf, o[dt]);
            }
        }
        __syncthreads();   // protect sK/sV before next chunk overwrites
    }

    // --- normalize and store [T, H*D] ---
#pragma unroll
    for (int dt = 0; dt < 8; ++dt)
#pragma unroll
        for (int r = 0; r < 8; ++r) {
            const int t = b * S_ + qtile * 16 + rowhi + r;
            attn[(size_t)t * (H_ * D_) + h * D_ + dt * 16 + mg] = o[dt][r] / lrow[r];
        }
}

// =====================================================================
extern "C" void kernel_launch(void* const* d_in, const int* in_sizes, int n_in,
                              void* d_out, int out_size, void* d_ws, size_t ws_size,
                              hipStream_t stream) {
    const float* hidden = (const float*)d_in[0];
    const float* cosp   = (const float*)d_in[1];
    const float* sinp   = (const float*)d_in[2];
    const float* qkv_w  = (const float*)d_in[3];
    const float* o_w    = (const float*)d_in[4];
    const float* kc_in  = (const float*)d_in[5];
    const float* vc_in  = (const float*)d_in[6];
    const int*   slots  = (const int*)d_in[8];
    const int*   btab   = (const int*)d_in[9];

    float* out = (float*)d_out;                                  // [T, HID]
    float* kc  = out + (size_t)T_ * HID_;                        // [NB, BS, HKV, D]
    float* vc  = kc + (size_t)NUM_BLOCKS_ * BLOCK_SIZE_ * HKV_ * D_;

    float* qkv  = (float*)d_ws;                                  // [T, 12288]
    float* attn = qkv + (size_t)T_ * QKV_N;                      // [T, 4096]

    const size_t cache_bytes = (size_t)NUM_BLOCKS_ * BLOCK_SIZE_ * HKV_ * D_ * sizeof(float);
    (void)hipMemcpyAsync(kc, kc_in, cache_bytes, hipMemcpyDeviceToDevice, stream);
    (void)hipMemcpyAsync(vc, vc_in, cache_bytes, hipMemcpyDeviceToDevice, stream);

    // 1) QKV projection: [4096,4096] x [4096,12288]
    gemm_bf16_wmma<<<dim3(QKV_N / 128, T_ / 128), 256, 0, stream>>>(
        hidden, qkv_w, qkv, T_, QKV_N, HID_, HID_, QKV_N, QKV_N);

    // 2) RoPE + cache scatter
    rope_scatter_kernel<<<T_, 128, 0, stream>>>(qkv, cosp, sinp, slots, kc, vc);

    // 3) Causal flash attention (4 q-tiles per block share K/V chunks)
    attn_kernel<<<dim3(S_ / 64, H_, B_), 128, 0, stream>>>(qkv, kc, vc, btab, attn);

    // 4) Output projection: [4096,4096] x [4096,4096]
    gemm_bf16_wmma<<<dim3(HID_ / 128, T_ / 128), 256, 0, stream>>>(
        attn, o_w, out, T_, HID_, HID_, H_ * D_, HID_, HID_);
}
